// EncoderForCoreMLExport_86449101734217
// MI455X (gfx1250) — matrix operand order repaired and checked by reading, not compiled, not run on hardware
//
#include <hip/hip_runtime.h>

// ---------------------------------------------------------------------------
// Bidirectional 2-layer LSTM (S=4096, IN=H=1024, batch=1) for MI455X/gfx1250.
//  * Input projections: bf16 WMMA GEMMs (v_wmma_f32_16x16x32_bf16, f32 accum),
//    register-blocked 16x64 per wave (A fragment reused over 4 WMMAs).
//  * Recurrent scans: persistent kernel, w_hh slice resident in LDS (64KB/WG,
//    CDNA5 has 320KB/WGP), grid-wide sense-reversing barrier per timestep.
// ---------------------------------------------------------------------------

#define S_LEN 4096
#define IN_SZ 1024
#define H_SZ  1024
#define G4    4096   // 4*H
#define H2    2048   // 2*H
#define NWG_SCAN 128 // 128 WGs x 8 h-elements = H

typedef __attribute__((ext_vector_type(16))) __bf16 v16bf;
typedef __attribute__((ext_vector_type(8)))  float  v8f;

__device__ __forceinline__ unsigned short f32_to_bf16(float f) {
  unsigned u = __float_as_uint(f);
  u += 0x7fffu + ((u >> 16) & 1u);           // round-to-nearest-even
  return (unsigned short)(u >> 16);
}
__device__ __forceinline__ float bf_lo(unsigned v) { return __uint_as_float(v << 16); }
__device__ __forceinline__ float bf_hi(unsigned v) { return __uint_as_float(v & 0xffff0000u); }
__device__ __forceinline__ float sigm(float x) { return 1.0f / (1.0f + __expf(-x)); }

// -------------------------------- fp32 -> bf16 ------------------------------
__global__ void cvt_f32_bf16(const float* __restrict__ src,
                             unsigned short* __restrict__ dst, int n) {
  int i = blockIdx.x * 256 + threadIdx.x;
  if (i < n) dst[i] = f32_to_bf16(src[i]);
}

// ------------------------------ WMMA GEMM -----------------------------------
// C[M,N] = A[M,K](bf16,row-major) @ W[N,K](bf16,row-major)^T + bias[N]
// Block = 256 threads = 8 waves arranged 4(M) x 2(N); each wave owns a 16x64
// C strip = 4 accumulators, sharing one A fragment per k-step (2.5 loads/WMMA).
// Fragment layouts used here (same contiguous k order in A and B):
//  A 16x32 bf16 : lane<16 -> row=lane, K = k0..k0+15 (contiguous)
//                 lane>=16 -> row=lane-16, K = k0+16..k0+31
//  B 32x16 bf16 : lane<16 -> col=lane, K = k0..k0+15 (contiguous)
//                 lane>=16 -> col=lane-16, K = k0+16..k0+31
//  C/D f32      : reg r, lane l -> M = r + (l>=16 ? 8:0), N = l&15
union Frag { uint4 u[2]; v16bf v; };

__device__ __forceinline__ v8f wmma_bf16(const Frag& a, const Frag& b, v8f c) {
  return __builtin_amdgcn_wmma_f32_16x16x32_bf16(false, a.v, false, b.v,
                                                 (short)0, c, false, false);
}

__global__ __launch_bounds__(256) void gemm_bias_wmma(
    const unsigned short* __restrict__ A, int lda,
    const unsigned short* __restrict__ W, int ldw,
    const float* __restrict__ bias,
    float* __restrict__ C, int ldc, int K)
{
  const int lane = threadIdx.x & 31;
  const int wave = threadIdx.x >> 5;                   // 0..7
  const int m0 = blockIdx.x * 64 + (wave & 3) * 16;    // 4 waves in M
  const int n0 = blockIdx.y * 128 + (wave >> 2) * 64;  // 2 waves in N, 64 wide

  const int am  = lane & 15;
  const int akb = (lane & 16) ? 16 : 0;
  const int bn  = lane & 15;
  const int bkb = (lane & 16) ? 16 : 0;

  const unsigned short* arow = A + (size_t)(m0 + am) * lda + akb;
  const unsigned short* wrow = W + (size_t)(n0 + bn) * ldw + bkb;
  const size_t wstep = (size_t)16 * ldw;   // next 16-column N subtile

  v8f acc0 = {}, acc1 = {}, acc2 = {}, acc3 = {};
  for (int k0 = 0; k0 < K; k0 += 32) {
    Frag fa;
    fa.u[0] = *reinterpret_cast<const uint4*>(arow + k0);
    fa.u[1] = *reinterpret_cast<const uint4*>(arow + k0 + 8);
    Frag fb0, fb1, fb2, fb3;
    fb0.u[0] = *reinterpret_cast<const uint4*>(wrow + k0);
    fb0.u[1] = *reinterpret_cast<const uint4*>(wrow + k0 + 8);
    fb1.u[0] = *reinterpret_cast<const uint4*>(wrow + wstep + k0);
    fb1.u[1] = *reinterpret_cast<const uint4*>(wrow + wstep + k0 + 8);
    fb2.u[0] = *reinterpret_cast<const uint4*>(wrow + 2 * wstep + k0);
    fb2.u[1] = *reinterpret_cast<const uint4*>(wrow + 2 * wstep + k0 + 8);
    fb3.u[0] = *reinterpret_cast<const uint4*>(wrow + 3 * wstep + k0);
    fb3.u[1] = *reinterpret_cast<const uint4*>(wrow + 3 * wstep + k0 + 8);
    acc0 = wmma_bf16(fa, fb0, acc0);
    acc1 = wmma_bf16(fa, fb1, acc1);
    acc2 = wmma_bf16(fa, fb2, acc2);
    acc3 = wmma_bf16(fa, fb3, acc3);
  }

  const int mo = (lane & 16) ? 8 : 0;
  float* crow = C + (size_t)(m0 + mo) * ldc + bn;
  const float bv0 = bias[n0 + bn];
  const float bv1 = bias[n0 + 16 + bn];
  const float bv2 = bias[n0 + 32 + bn];
  const float bv3 = bias[n0 + 48 + bn];
#pragma unroll
  for (int r = 0; r < 8; ++r) {
    crow[(size_t)r * ldc + n0]      = acc0[r] + bv0;
    crow[(size_t)r * ldc + n0 + 16] = acc1[r] + bv1;
    crow[(size_t)r * ldc + n0 + 32] = acc2[r] + bv2;
    crow[(size_t)r * ldc + n0 + 48] = acc3[r] + bv3;
  }
}

// --------------------------- persistent LSTM scan ---------------------------
// Each of 128 WGs owns 8 h-elements (jbase..jbase+7) -> 32 gate rows of w_hh,
// held in LDS as bf16 (32 x 1024 x 2B = 64KB). Per step: dot-products (8 lanes
// per row, shfl reduce), pointwise gate math on threads 0..7, ping-pong h in
// global memory, grid-wide barrier.
__global__ void scan_init(unsigned short* hbuf, unsigned* sync_count,
                          unsigned* sync_gen) {
  int i = blockIdx.x * 256 + threadIdx.x;
  if (i < 2 * H_SZ) hbuf[i] = 0;
  if (i == 0) { *sync_count = 0u; *sync_gen = 0u; }
}

__global__ __launch_bounds__(256) void lstm_scan(
    const unsigned short* __restrict__ whh,   // [4H, H] bf16
    const float* __restrict__ xw,             // [S, 4H] f32 (x@Wih^T + b)
    unsigned short* __restrict__ hbuf,        // [2, H] bf16 ping-pong (zeroed)
    float* __restrict__ hout,                 // [H] f32, final h after loop
    unsigned short* __restrict__ hs_out,      // [S, 2H] bf16 history or null
    int hs_col, int reverse,
    unsigned* __restrict__ sync_count, unsigned* __restrict__ sync_gen)
{
  __shared__ unsigned short whh_s[32 * H_SZ];   // 64 KB resident weight slice
  __shared__ unsigned short h_s[H_SZ];          // 2 KB staged h
  __shared__ float gate_s[32];

  const int t = threadIdx.x;
  const int jbase = blockIdx.x * 8;

  // Preload this WG's 32 gate rows (i,f,g,o x 8 j's) into LDS, 16B chunks.
  for (int idx = t; idx < 32 * (H_SZ / 8); idx += 256) {
    int r = idx >> 7;            // local row 0..31 (128 uint4 per row)
    int cs = idx & 127;
    int q = r >> 3, jj = r & 7;  // gate, j-within-WG
    const uint4* src =
        reinterpret_cast<const uint4*>(whh + (size_t)(q * H_SZ + jbase + jj) * H_SZ) + cs;
    reinterpret_cast<uint4*>(whh_s)[(size_t)r * 128 + cs] = *src;
  }

  const int row = t >> 3;   // 0..31  (= gate*8 + jj)
  const int s8  = t & 7;    // k-slice 0..7 (128 elements each)
  float c_state = 0.0f;
  __syncthreads();

  for (int step = 0; step < S_LEN; ++step) {
    const int ti = reverse ? (S_LEN - 1 - step) : step;
    const unsigned short* hcur = hbuf + (step & 1) * H_SZ;
    unsigned short*       hnxt = hbuf + ((step + 1) & 1) * H_SZ;

    __threadfence();   // acquire h written by other WGs last step
    for (int i = t; i < H_SZ / 8; i += 256)
      reinterpret_cast<uint4*>(h_s)[i] = reinterpret_cast<const uint4*>(hcur)[i];
    __syncthreads();

    // partial dot over k = s8*128 .. +127 (bf16 weights * bf16 h, f32 accum)
    const uint4* wp = reinterpret_cast<const uint4*>(whh_s + row * H_SZ + s8 * 128);
    const uint4* hp = reinterpret_cast<const uint4*>(h_s + s8 * 128);
    float sum = 0.0f;
#pragma unroll 4
    for (int i = 0; i < 16; ++i) {
      uint4 w4 = wp[i], h4 = hp[i];
      sum += bf_lo(w4.x) * bf_lo(h4.x) + bf_hi(w4.x) * bf_hi(h4.x);
      sum += bf_lo(w4.y) * bf_lo(h4.y) + bf_hi(w4.y) * bf_hi(h4.y);
      sum += bf_lo(w4.z) * bf_lo(h4.z) + bf_hi(w4.z) * bf_hi(h4.z);
      sum += bf_lo(w4.w) * bf_lo(h4.w) + bf_hi(w4.w) * bf_hi(h4.w);
    }
    sum += __shfl_xor(sum, 1, 32);
    sum += __shfl_xor(sum, 2, 32);
    sum += __shfl_xor(sum, 4, 32);
    if (s8 == 0) gate_s[row] = sum;
    __syncthreads();

    if (t < 8) {                 // one thread per owned h element
      const int j = jbase + t;
      const float* xwrow = xw + (size_t)ti * G4;
      float gi = xwrow[j]            + gate_s[t];
      float gf = xwrow[H_SZ + j]     + gate_s[8 + t];
      float gg = xwrow[2 * H_SZ + j] + gate_s[16 + t];
      float go = xwrow[3 * H_SZ + j] + gate_s[24 + t];
      c_state = sigm(gf) * c_state + sigm(gi) * tanhf(gg);
      float h = sigm(go) * tanhf(c_state);
      unsigned short hb = f32_to_bf16(h);
      hnxt[j] = hb;
      hout[j] = h;
      if (hs_out) hs_out[(size_t)ti * H2 + hs_col + j] = hb;
    }
    __threadfence();             // release h before grid barrier
    __syncthreads();

    if (t == 0) {                // sense-reversing grid barrier
      volatile unsigned* vg = (volatile unsigned*)sync_gen;
      unsigned prev = atomicAdd(sync_count, 1u);
      if (prev == NWG_SCAN - 1) {
        *(volatile unsigned*)sync_count = 0u;
        __threadfence();
        *vg = (unsigned)(step + 1);
      } else {
        while (*vg < (unsigned)(step + 1)) __builtin_amdgcn_s_sleep(2);
      }
    }
    __syncthreads();
  }
}

// ------------------------------- finalize -----------------------------------
__global__ void finalize(const float* __restrict__ hf, const float* __restrict__ hr,
                         float* __restrict__ out, int out_size) {
  int i = blockIdx.x * 256 + threadIdx.x;
  if (i < H_SZ)                    out[i] = hf[i];
  else if (i < 2 * H_SZ)           out[i] = hr[i - H_SZ];
  else if (i < out_size)           out[i] = 0.0f;   // tuple's scalar 0.0
}

// ------------------------------- launcher -----------------------------------
extern "C" void kernel_launch(void* const* d_in, const int* in_sizes, int n_in,
                              void* d_out, int out_size, void* d_ws, size_t ws_size,
                              hipStream_t stream) {
  (void)in_sizes; (void)n_in; (void)ws_size;
  const float* x     = (const float*)d_in[0];
  const float* wih0f = (const float*)d_in[1];
  const float* whh0f = (const float*)d_in[2];
  const float* b0f   = (const float*)d_in[3];
  const float* wih0r = (const float*)d_in[4];
  const float* whh0r = (const float*)d_in[5];
  const float* b0r   = (const float*)d_in[6];
  const float* wih1f = (const float*)d_in[7];
  const float* whh1f = (const float*)d_in[8];
  const float* b1f   = (const float*)d_in[9];
  const float* wih1r = (const float*)d_in[10];
  const float* whh1r = (const float*)d_in[11];
  const float* b1r   = (const float*)d_in[12];

  char* ws = (char*)d_ws;
  size_t off = 0;
  auto take = [&](size_t bytes) -> char* {
    char* p = ws + off;
    off += (bytes + 255) & ~(size_t)255;
    return p;
  };

  unsigned short* x_b     = (unsigned short*)take((size_t)S_LEN * IN_SZ * 2);
  unsigned short* wih0f_b = (unsigned short*)take((size_t)G4 * IN_SZ * 2);
  unsigned short* wih0r_b = (unsigned short*)take((size_t)G4 * IN_SZ * 2);
  unsigned short* wih1f_b = (unsigned short*)take((size_t)G4 * H2 * 2);
  unsigned short* wih1r_b = (unsigned short*)take((size_t)G4 * H2 * 2);
  unsigned short* whh0f_b = (unsigned short*)take((size_t)G4 * H_SZ * 2);
  unsigned short* whh0r_b = (unsigned short*)take((size_t)G4 * H_SZ * 2);
  unsigned short* whh1f_b = (unsigned short*)take((size_t)G4 * H_SZ * 2);
  unsigned short* whh1r_b = (unsigned short*)take((size_t)G4 * H_SZ * 2);
  float*          xw_f    = (float*)take((size_t)S_LEN * G4 * 4);
  float*          xw_r    = (float*)take((size_t)S_LEN * G4 * 4);
  unsigned short* x1_b    = (unsigned short*)take((size_t)S_LEN * H2 * 2);
  unsigned short* hbuf    = (unsigned short*)take(2 * H_SZ * 2);
  float*          hfin_f  = (float*)take(H_SZ * 4);
  float*          hfin_r  = (float*)take(H_SZ * 4);
  unsigned*       sc      = (unsigned*)take(256);
  unsigned*       sg      = sc + 8;

  auto cvt = [&](const float* s, unsigned short* d, int n) {
    cvt_f32_bf16<<<(n + 255) / 256, 256, 0, stream>>>(s, d, n);
  };
  cvt(x,     x_b,     S_LEN * IN_SZ);
  cvt(wih0f, wih0f_b, G4 * IN_SZ);
  cvt(wih0r, wih0r_b, G4 * IN_SZ);
  cvt(wih1f, wih1f_b, G4 * H2);
  cvt(wih1r, wih1r_b, G4 * H2);
  cvt(whh0f, whh0f_b, G4 * H_SZ);
  cvt(whh0r, whh0r_b, G4 * H_SZ);
  cvt(whh1f, whh1f_b, G4 * H_SZ);
  cvt(whh1r, whh1r_b, G4 * H_SZ);

  dim3 ggrid(S_LEN / 64, G4 / 128);
  // layer 0 input projections
  gemm_bias_wmma<<<ggrid, 256, 0, stream>>>(x_b, IN_SZ, wih0f_b, IN_SZ, b0f, xw_f, G4, IN_SZ);
  gemm_bias_wmma<<<ggrid, 256, 0, stream>>>(x_b, IN_SZ, wih0r_b, IN_SZ, b0r, xw_r, G4, IN_SZ);

  // layer 0 scans (write h history as bf16 into x1 = [hf | hr])
  scan_init<<<8, 256, 0, stream>>>(hbuf, sc, sg);
  lstm_scan<<<NWG_SCAN, 256, 0, stream>>>(whh0f_b, xw_f, hbuf, hfin_f, x1_b, 0,    0, sc, sg);
  scan_init<<<8, 256, 0, stream>>>(hbuf, sc, sg);
  lstm_scan<<<NWG_SCAN, 256, 0, stream>>>(whh0r_b, xw_r, hbuf, hfin_r, x1_b, H_SZ, 1, sc, sg);

  // layer 1 input projections (K = 2H), reuse xw buffers
  gemm_bias_wmma<<<ggrid, 256, 0, stream>>>(x1_b, H2, wih1f_b, H2, b1f, xw_f, G4, H2);
  gemm_bias_wmma<<<ggrid, 256, 0, stream>>>(x1_b, H2, wih1r_b, H2, b1r, xw_r, G4, H2);

  // layer 1 scans: only final hidden states needed
  scan_init<<<8, 256, 0, stream>>>(hbuf, sc, sg);
  lstm_scan<<<NWG_SCAN, 256, 0, stream>>>(whh1f_b, xw_f, hbuf, hfin_f, (unsigned short*)nullptr, 0, 0, sc, sg);
  scan_init<<<8, 256, 0, stream>>>(hbuf, sc, sg);
  lstm_scan<<<NWG_SCAN, 256, 0, stream>>>(whh1r_b, xw_r, hbuf, hfin_r, (unsigned short*)nullptr, 0, 1, sc, sg);

  finalize<<<(out_size + 255) / 256, 256, 0, stream>>>(hfin_f, hfin_r, (float*)d_out, out_size);
}
